// BARTLongformerModel_46918222741897
// MI455X (gfx1250) — matrix operand order repaired and checked by reading, not compiled, not run on hardware
//
#include <hip/hip_runtime.h>
#include <stdint.h>

#define B_  2
#define S_  4096
#define V_  16384
#define D_  768
#define H_  12
#define L_  4
#define W_  256
#define HD_ 64
#define M_  (B_ * S_)   // 8192 rows

typedef __attribute__((ext_vector_type(16))) __bf16 v16bf;
typedef __attribute__((ext_vector_type(8)))  float  v8f;

union FragBF {
    v16bf v;
    uint4 q[2];
    unsigned short u[16];
};

__device__ __forceinline__ unsigned short f32_to_bf16(float f) {
    union { float f; unsigned int u; } c; c.f = f;
    unsigned int u = c.u;
    unsigned int r = u + 0x7FFFu + ((u >> 16) & 1u);   // round-to-nearest-even
    return (unsigned short)(r >> 16);
}

// Async DMA: global (per-lane 16B) -> LDS, tracked by ASYNCcnt.
__device__ __forceinline__ void gmem_to_lds_async_b128(
    unsigned short* lds, const unsigned short* g)
{
    unsigned int loff = (unsigned int)(unsigned long long)lds;  // LDS byte address
    asm volatile("global_load_async_to_lds_b128 %0, %1, off"
                 :: "v"(loff), "v"(g) : "memory");
}

template <int N>
__device__ __forceinline__ void wait_asynccnt() {
    asm volatile("s_wait_asynccnt %0" :: "i"(N) : "memory");
}

// ---------------------------------------------------------------------------
// Embedding lookup + sinusoidal positional encoding -> bf16 activations
// ---------------------------------------------------------------------------
__global__ __launch_bounds__(256) void embed_pe_kernel(
    const int* __restrict__ x, const float* __restrict__ tbl,
    unsigned short* __restrict__ h)
{
    int bs  = blockIdx.x;            // 0 .. B*S-1
    int s   = bs % S_;
    int tok = x[bs];
    const float* row = tbl + (size_t)tok * D_;
    float pos = (float)s;
    #pragma unroll
    for (int r = 0; r < D_ / 256; ++r) {
        int d  = r * 256 + threadIdx.x;
        int i2 = (d >> 1) * 2;                                     // 2i
        float div = __expf((float)i2 * (-9.210340371976184f / (float)D_)); // -ln(1e4)/D
        float ang = pos * div;
        float pe  = (d & 1) ? __cosf(ang) : __sinf(ang);
        h[(size_t)bs * D_ + d] = f32_to_bf16(row[d] + pe);
    }
}

// ---------------------------------------------------------------------------
// Convert fp32 [K][N] weight to bf16 transposed [N][K]
// ---------------------------------------------------------------------------
__global__ __launch_bounds__(256) void convT_kernel(
    const float* __restrict__ in, unsigned short* __restrict__ out, int K, int N)
{
    int idx = blockIdx.x * 256 + threadIdx.x;
    if (idx >= K * N) return;
    int n = idx / K, k = idx % K;
    out[idx] = f32_to_bf16(in[(size_t)k * N + n]);
}

// ---------------------------------------------------------------------------
// bf16 WMMA GEMM: C[M,N] = (A[M,K] * B[K,N] + bias[N]) * alpha
// B supplied transposed: BT[N][K].  128x128 block tile, BK=64, 8 waves.
// Double-buffered async global->LDS pipeline (8 ASYNCcnt ops per stage,
// 16 WMMAs per stage between barriers).
// ---------------------------------------------------------------------------
template <bool OUT_F32>
__global__ __launch_bounds__(256) void gemm_bf16_kernel(
    const unsigned short* __restrict__ A,    // [M,K] bf16
    const unsigned short* __restrict__ BT,   // [N,K] bf16
    const float* __restrict__ bias,          // [N]
    void* __restrict__ Cout, int M, int N, int K, float alpha)
{
    __shared__ __align__(16) unsigned short As[2][128 * 64];   // 2 x 16 KB
    __shared__ __align__(16) unsigned short Bs[2][128 * 64];   // 2 x 16 KB

    const int bn   = blockIdx.x * 128;
    const int bm   = blockIdx.y * 128;
    const int tid  = threadIdx.x;
    const int wave = tid >> 5;
    const int lane = tid & 31;
    const int wm   = (wave & 3) * 32;   // 4 waves along M
    const int wn   = (wave >> 2) * 64;  // 2 waves along N
    const int lk   = lane & 15;
    const int hlf  = lane >> 4;
    const int arow = tid >> 1;          // 128 rows, 2 threads/row
    const int acol = (tid & 1) * 32;    // 32 bf16 (64 B) per thread

    v8f acc[2][4] = {};

    const unsigned short* gA = A  + (size_t)(bm + arow) * K + acol;
    const unsigned short* gB = BT + (size_t)(bn + arow) * K + acol;

    auto issue_tile = [&](int buf, int k0) {
        unsigned short* la = &As[buf][arow * 64 + acol];
        unsigned short* lb = &Bs[buf][arow * 64 + acol];
        gmem_to_lds_async_b128(la,      gA + k0);
        gmem_to_lds_async_b128(la + 8,  gA + k0 + 8);
        gmem_to_lds_async_b128(la + 16, gA + k0 + 16);
        gmem_to_lds_async_b128(la + 24, gA + k0 + 24);
        gmem_to_lds_async_b128(lb,      gB + k0);
        gmem_to_lds_async_b128(lb + 8,  gB + k0 + 8);
        gmem_to_lds_async_b128(lb + 16, gB + k0 + 16);
        gmem_to_lds_async_b128(lb + 24, gB + k0 + 24);
    };

    issue_tile(0, 0);
    const int KT = K / 64;
    for (int kt = 0; kt < KT; ++kt) {
        if (kt + 1 < KT) {
            issue_tile((kt + 1) & 1, (kt + 1) * 64);
            if (kt + 2 < KT)
                __builtin_prefetch((const char*)(gA + (kt + 2) * 64), 0, 3);
            wait_asynccnt<8>();   // current stage's 8 async ops retired
        } else {
            wait_asynccnt<0>();
        }
        __syncthreads();

        const unsigned short* Ab = As[kt & 1];
        const unsigned short* Bb = Bs[kt & 1];
        #pragma unroll
        for (int kk2 = 0; kk2 < 2; ++kk2) {            // two 32-deep K halves
            FragBF af[2], bfg[4];
            #pragma unroll
            for (int i = 0; i < 2; ++i) {
                const unsigned short* p = Ab + (wm + i * 16 + lk) * 64 + kk2 * 32;
                af[i].q[0] = *(const uint4*)(p + hlf * 8);       // K = half*8 .. +7
                af[i].q[1] = *(const uint4*)(p + 16 + hlf * 8);  // K = 16+half*8 .. +7
            }
            #pragma unroll
            for (int j = 0; j < 4; ++j) {
                const unsigned short* p = Bb + (wn + j * 16 + lk) * 64 + kk2 * 32;
                bfg[j].q[0] = *(const uint4*)(p + hlf * 8);      // K = half*8 .. +7
                bfg[j].q[1] = *(const uint4*)(p + 16 + hlf * 8); // K = 16+half*8 .. +7
            }
            #pragma unroll
            for (int i = 0; i < 2; ++i)
                #pragma unroll
                for (int j = 0; j < 4; ++j)
                    acc[i][j] = __builtin_amdgcn_wmma_f32_16x16x32_bf16(
                        false, af[i].v, false, bfg[j].v, (short)0, acc[i][j], false, false);
        }
        __syncthreads();   // reads done before this buffer is overwritten
    }

    #pragma unroll
    for (int i = 0; i < 2; ++i)
        #pragma unroll
        for (int j = 0; j < 4; ++j) {
            int row0 = bm + wm + i * 16 + hlf * 8;
            int col  = bn + wn + j * 16 + lk;
            float bv = bias[col];
            #pragma unroll
            for (int e = 0; e < 8; ++e) {
                float v   = (acc[i][j][e] + bv) * alpha;
                size_t id = (size_t)(row0 + e) * N + col;
                if (OUT_F32) ((float*)Cout)[id] = v;
                else         ((unsigned short*)Cout)[id] = f32_to_bf16(v);
            }
        }
}

// ---------------------------------------------------------------------------
// Banded (sliding-window) attention, flash-style online softmax.
// One block per (b, h, chunk of 256 queries); 8 waves x 32 queries.
// Wave w only touches key tiles kt in [w, w+16] (band coverage).
// ---------------------------------------------------------------------------
__global__ __launch_bounds__(256) void band_attn_kernel(
    const unsigned short* __restrict__ Q,
    const unsigned short* __restrict__ Km,
    const unsigned short* __restrict__ Vm,
    unsigned short* __restrict__ O)
{
    __shared__ __align__(16) unsigned short Ks [32 * 64];       // [key][hd]
    __shared__ __align__(16) unsigned short Vts[64 * 32];       // [hd][key] (transposed)
    __shared__ __align__(16) unsigned short Ps [8 * 32 * 32];   // per-wave P bounce

    const int c    = blockIdx.x;  // chunk 0..15
    const int h    = blockIdx.y;
    const int b    = blockIdx.z;
    const int tid  = threadIdx.x;
    const int wave = tid >> 5;
    const int lane = tid & 31;
    const int lk   = lane & 15;
    const int hlf  = lane >> 4;

    // Persistent Q fragments (2 row tiles x 2 K-steps over HD=64)
    FragBF aq[2][2];
    #pragma unroll
    for (int qm = 0; qm < 2; ++qm) {
        const unsigned short* qrow =
            Q + ((size_t)(b * S_ + c * W_ + wave * 32 + qm * 16 + lk)) * D_ + h * HD_;
        #pragma unroll
        for (int kk = 0; kk < 2; ++kk) {
            aq[qm][kk].q[0] = *(const uint4*)(qrow + kk * 32 + hlf * 8);
            aq[qm][kk].q[1] = *(const uint4*)(qrow + kk * 32 + 16 + hlf * 8);
        }
    }

    v8f acc[2][4] = {};
    float mrow[2][8], lrow[2][8];
    #pragma unroll
    for (int qm = 0; qm < 2; ++qm)
        #pragma unroll
        for (int e = 0; e < 8; ++e) { mrow[qm][e] = -1e30f; lrow[qm][e] = 0.f; }

    const int r    = tid >> 3;        // key row 0..31 within tile
    const int hseg = (tid & 7) * 8;   // 8 bf16 of head dim per thread

    for (int kt = 0; kt < 24; ++kt) {              // 24 * 32 = 768 = 3W halo keys
        int j = (c - 1) * W_ + kt * 32 + r;
        __syncthreads();
        {
            uint4 kv = {0, 0, 0, 0}, vv = {0, 0, 0, 0};
            if (j >= 0 && j < S_) {
                size_t base = ((size_t)(b * S_ + j)) * D_ + h * HD_ + hseg;
                kv = *(const uint4*)(Km + base);
                vv = *(const uint4*)(Vm + base);
            }
            *(uint4*)(Ks + r * 64 + hseg) = kv;
            union { uint4 q; unsigned short u[8]; } vu; vu.q = vv;
            #pragma unroll
            for (int e = 0; e < 8; ++e) Vts[(hseg + e) * 32 + r] = vu.u[e];
        }
        __syncthreads();

        // Band coverage: this wave's 32 queries only see tiles [wave, wave+16].
        const bool active = (kt >= wave) && (kt <= wave + 16);

        if (active) {
            // S = Q * K^T  (B-frag = rows of K, since BT == K)
            v8f sc[2][2];
            #pragma unroll
            for (int qm = 0; qm < 2; ++qm)
                #pragma unroll
                for (int kn = 0; kn < 2; ++kn) {
                    v8f s = {};
                    #pragma unroll
                    for (int kk = 0; kk < 2; ++kk) {
                        FragBF bk;
                        const unsigned short* p = Ks + (kn * 16 + lk) * 64 + kk * 32;
                        bk.q[0] = *(const uint4*)(p + hlf * 8);
                        bk.q[1] = *(const uint4*)(p + 16 + hlf * 8);
                        s = __builtin_amdgcn_wmma_f32_16x16x32_bf16(
                            false, aq[qm][kk].v, false, bk.v, (short)0, s, false, false);
                    }
                    sc[qm][kn] = s;
                }

            // Online softmax update per query row
            #pragma unroll
            for (int qm = 0; qm < 2; ++qm) {
                float fsc[8];
                #pragma unroll
                for (int e = 0; e < 8; ++e) {
                    int i  = c * W_ + wave * 32 + qm * 16 + e + 8 * hlf;
                    int j0 = (c - 1) * W_ + kt * 32 + lk;
                    int j1 = j0 + 16;
                    bool m0 = (j0 >= i - W_) && (j0 <= i + W_) && (j0 >= 0) && (j0 < S_);
                    bool m1 = (j1 >= i - W_) && (j1 <= i + W_) && (j1 >= 0) && (j1 < S_);
                    float s0 = m0 ? sc[qm][0][e] : -1e30f;
                    float s1 = m1 ? sc[qm][1][e] : -1e30f;
                    float mx = fmaxf(s0, s1);
                    #pragma unroll
                    for (int d = 1; d < 16; d <<= 1) mx = fmaxf(mx, __shfl_xor(mx, d, 32));
                    float mold = mrow[qm][e];
                    float mnew = fmaxf(mold, mx);
                    float f    = __expf(mold - mnew);
                    float p0   = m0 ? __expf(s0 - mnew) : 0.f;
                    float p1   = m1 ? __expf(s1 - mnew) : 0.f;
                    float rs   = p0 + p1;
                    #pragma unroll
                    for (int d = 1; d < 16; d <<= 1) rs += __shfl_xor(rs, d, 32);
                    lrow[qm][e] = lrow[qm][e] * f + rs;
                    mrow[qm][e] = mnew;
                    fsc[e] = f;
                    int rloc = qm * 16 + e + 8 * hlf;
                    Ps[wave * 1024 + rloc * 32 + lk]      = f32_to_bf16(p0);
                    Ps[wave * 1024 + rloc * 32 + 16 + lk] = f32_to_bf16(p1);
                }
                #pragma unroll
                for (int on = 0; on < 4; ++on)
                    #pragma unroll
                    for (int e = 0; e < 8; ++e)
                        acc[qm][on][e] *= fsc[e];
            }
        }
        __syncthreads();   // uniform barrier (also orders per-wave Ps stores)

        if (active) {
            // O += P * V  (A-frag from Ps, B-frag = rows of Vt)
            #pragma unroll
            for (int qm = 0; qm < 2; ++qm) {
                FragBF ap;
                const unsigned short* pp = Ps + wave * 1024 + (qm * 16 + lk) * 32;
                ap.q[0] = *(const uint4*)(pp + hlf * 8);
                ap.q[1] = *(const uint4*)(pp + 16 + hlf * 8);
                #pragma unroll
                for (int on = 0; on < 4; ++on) {
                    FragBF bv;
                    const unsigned short* vp = Vts + (on * 16 + lk) * 32;
                    bv.q[0] = *(const uint4*)(vp + hlf * 8);
                    bv.q[1] = *(const uint4*)(vp + 16 + hlf * 8);
                    acc[qm][on] = __builtin_amdgcn_wmma_f32_16x16x32_bf16(
                        false, ap.v, false, bv.v, (short)0, acc[qm][on], false, false);
                }
            }
        }
    }

    // Normalize and write bf16 output (head-major [S, H*HD] layout)
    #pragma unroll
    for (int qm = 0; qm < 2; ++qm)
        #pragma unroll
        for (int on = 0; on < 4; ++on)
            #pragma unroll
            for (int e = 0; e < 8; ++e) {
                int rloc  = wave * 32 + qm * 16 + e + 8 * hlf;
                float inv = 1.f / lrow[qm][e];
                float v   = acc[qm][on][e] * inv;
                O[((size_t)(b * S_ + c * W_ + rloc)) * D_ + h * HD_ + on * 16 + lk] =
                    f32_to_bf16(v);
            }
}

// ---------------------------------------------------------------------------
extern "C" void kernel_launch(void* const* d_in, const int* in_sizes, int n_in,
                              void* d_out, int out_size, void* d_ws, size_t ws_size,
                              hipStream_t stream)
{
    const int*   x    = (const int*)  d_in[0];
    const float* tbl  = (const float*)d_in[1];
    const float* Wq   = (const float*)d_in[2];
    const float* bq   = (const float*)d_in[3];
    const float* Wk   = (const float*)d_in[4];
    const float* bk   = (const float*)d_in[5];
    const float* Wv   = (const float*)d_in[6];
    const float* bv   = (const float*)d_in[7];
    const float* Wout = (const float*)d_in[8];
    const float* bout = (const float*)d_in[9];
    float* out = (float*)d_out;

    char* ws = (char*)d_ws;
    const size_t hBytes = (size_t)M_ * D_ * 2;        // 12.58 MB
    const size_t wBytes = (size_t)L_ * D_ * D_ * 2;   // 4.72 MB
    const size_t oBytes = (size_t)D_ * V_ * 2;        // 25.17 MB
    unsigned short* h0  = (unsigned short*)ws;  ws += hBytes;
    unsigned short* h1  = (unsigned short*)ws;  ws += hBytes;
    unsigned short* qb  = (unsigned short*)ws;  ws += hBytes;
    unsigned short* kb  = (unsigned short*)ws;  ws += hBytes;
    unsigned short* vb  = (unsigned short*)ws;  ws += hBytes;
    unsigned short* WqT = (unsigned short*)ws;  ws += wBytes;
    unsigned short* WkT = (unsigned short*)ws;  ws += wBytes;
    unsigned short* WvT = (unsigned short*)ws;  ws += wBytes;
    unsigned short* WoT = (unsigned short*)ws;  ws += oBytes;

    // 1) Embedding + PE
    embed_pe_kernel<<<M_, 256, 0, stream>>>(x, tbl, h0);

    // 2) Weight convert + transpose to bf16 [N][K]
    const int nDD = D_ * D_;
    const int gDD = (nDD + 255) / 256;
    for (int l = 0; l < L_; ++l) {
        convT_kernel<<<gDD, 256, 0, stream>>>(Wq + (size_t)l * nDD, WqT + (size_t)l * nDD, D_, D_);
        convT_kernel<<<gDD, 256, 0, stream>>>(Wk + (size_t)l * nDD, WkT + (size_t)l * nDD, D_, D_);
        convT_kernel<<<gDD, 256, 0, stream>>>(Wv + (size_t)l * nDD, WvT + (size_t)l * nDD, D_, D_);
    }
    const int nDV = D_ * V_;
    convT_kernel<<<(nDV + 255) / 256, 256, 0, stream>>>(Wout, WoT, D_, V_);

    // 3) Layer stack
    const float scale = 0.125f;   // 1/sqrt(HD)
    unsigned short* hc = h0;
    unsigned short* hn = h1;
    dim3 gQ(D_ / 128, M_ / 128);
    dim3 gA(S_ / W_, H_, B_);
    for (int l = 0; l < L_; ++l) {
        gemm_bf16_kernel<false><<<gQ, 256, 0, stream>>>(hc, WqT + (size_t)l * nDD, bq + l * D_, qb, M_, D_, D_, scale);
        gemm_bf16_kernel<false><<<gQ, 256, 0, stream>>>(hc, WkT + (size_t)l * nDD, bk + l * D_, kb, M_, D_, D_, 1.0f);
        gemm_bf16_kernel<false><<<gQ, 256, 0, stream>>>(hc, WvT + (size_t)l * nDD, bv + l * D_, vb, M_, D_, D_, 1.0f);
        band_attn_kernel<<<gA, 256, 0, stream>>>(qb, kb, vb, hn);
        unsigned short* t = hc; hc = hn; hn = t;
    }

    // 4) Output projection -> fp32 logits
    dim3 gO(V_ / 128, M_ / 128);
    gemm_bf16_kernel<true><<<gO, 256, 0, stream>>>(hc, WoT, bout, out, M_, V_, D_, 1.0f);
}
